// LstmRL_2018634629837
// MI455X (gfx1250) — matrix operand order repaired and checked by reading, not compiled, not run on hardware
//
#include <hip/hip_runtime.h>
#include <hip/hip_bf16.h>

// ---------------------------------------------------------------------------
// GraphRNN LSTM forward loss on MI455X (gfx1250).
// Persistent kernel: 128 blocks x 256 threads (8 waves), each block owns 32
// batch rows for the entire 64-step recurrence. Gates computed with
// v_wmma_f32_16x16x32_bf16; weights pre-converted to bf16 in d_ws (L2-resident).
// ---------------------------------------------------------------------------

#define HID   128
#define GATES 512
#define KDIM  256     // [x | h]
#define BT    32      // batch rows per block
#define XHP   264     // xh row pitch in bf16 elems (pad 8 for LDS banking)
#define GP    516     // gates row pitch in floats (pad 4 for LDS banking)
#define NFIELDS 64

typedef __attribute__((ext_vector_type(16))) __bf16 v16bf;
typedef __attribute__((ext_vector_type(8)))  float  v8f;

struct Ptrs {
    const float* soft[NFIELDS];
    const float* emb[NFIELDS];
};

__device__ __forceinline__ unsigned short f2bf(float f) {
    unsigned int u = __float_as_uint(f);
    unsigned int r = u + 0x7FFFu + ((u >> 16) & 1u);   // round-to-nearest-even
    return (unsigned short)(r >> 16);
}
__device__ __forceinline__ float bf2f(unsigned short s) {
    return __uint_as_float(((unsigned int)s) << 16);
}
__device__ __forceinline__ float sigf(float x) {
    return 1.0f / (1.0f + __expf(-x));
}

// Assemble a WMMA A-fragment from two 16B LDS loads (contiguous k = 16h + s, same order as B).
__device__ __forceinline__ v16bf load_frag16(const unsigned short* p0,
                                             const unsigned short* p1) {
    union { struct { uint4 lo, hi; } u; v16bf v; } cv;
    cv.u.lo = *(const uint4*)p0;
    cv.u.hi = *(const uint4*)p1;
    return cv.v;
}

// ---------------------------------------------------------------------------
// Prep: concat + convert weights to bf16: Wcat[l][g][k], k<128 -> w_ih, else w_hh.
// Also zeroes the scalar output.
// ---------------------------------------------------------------------------
__global__ void lstm_prep(const float* __restrict__ wih0, const float* __restrict__ wih1,
                          const float* __restrict__ whh0, const float* __restrict__ whh1,
                          unsigned short* __restrict__ Wcat, float* __restrict__ out)
{
    int idx = blockIdx.x * 256 + threadIdx.x;
    if (idx == 0) out[0] = 0.0f;
    if (idx >= 2 * GATES * KDIM) return;
    int l = idx / (GATES * KDIM);
    int r = idx % (GATES * KDIM);
    int g = r / KDIM, k = r % KDIM;
    const float* wih = l ? wih1 : wih0;
    const float* whh = l ? whh1 : whh0;
    float v = (k < HID) ? wih[g * HID + k] : whh[g * HID + (k - HID)];
    Wcat[idx] = f2bf(v);
}

// ---------------------------------------------------------------------------
// Main persistent kernel.
// ---------------------------------------------------------------------------
__global__ __launch_bounds__(256) void lstm_main(
    const float* __restrict__ adj,      // [4096][1][33][33]
    const int*   __restrict__ nl,       // [4096][1][33]
    const float* __restrict__ gemb,     // [1][128]
    const unsigned short* __restrict__ Wcat,  // bf16 [2][512][256]
    Ptrs P, float* __restrict__ out)
{
    __shared__ __align__(32) unsigned short xh0[BT * XHP];  // [x|h] layer0, bf16
    __shared__ __align__(32) unsigned short xh1[BT * XHP];  // [x|h] layer1, bf16
    __shared__ float c0[BT * HID];
    __shared__ float c1[BT * HID];
    __shared__ float gates[BT * GP];
    __shared__ float logits[BT * 32];
    __shared__ float decf[BT * 32];
    __shared__ int   decl[BT];
    __shared__ float dnm[BT];
    __shared__ float rowAcc[BT];

    const int tid   = threadIdx.x;
    const int wave  = tid >> 5;
    const int lane  = tid & 31;
    const int laneM = lane & 15;
    const int laneH = lane >> 4;
    const int bg0   = blockIdx.x * BT;

    // ---- init state ----
    for (int e = tid; e < BT * HID; e += 256) {
        int b = e >> 7, j = e & 127;
        c0[e] = 0.0f; c1[e] = 0.0f;
        xh0[b * XHP + j]       = f2bf(gemb[j]);  // x = g_emb broadcast
        xh0[b * XHP + HID + j] = 0;              // h0 = 0
        xh1[b * XHP + HID + j] = 0;              // h1 = 0
    }
    if (tid < BT) rowAcc[tid] = 0.0f;
    __syncthreads();

    for (int fi = 0; fi < NFIELDS; ++fi) {
        const int node  = (fi >> 1) + 1;
        const int isAdj = fi & 1;
        const int t     = isAdj ? node : 5;

        // =============== two LSTM layers ===============
        for (int layer = 0; layer < 2; ++layer) {
            const unsigned short* W  = Wcat + layer * GATES * KDIM;
            const unsigned short* xh = layer ? xh1 : xh0;

            // --- WMMA GEMM: gates[32][512] = xh[32][256] @ W^T ---
            const int nBase = wave * 64;       // 8 waves x 64 gate cols
            for (int nt = 0; nt < 4; ++nt) {
                const int n0 = nBase + nt * 16;
                v8f acc0 = {};
                v8f acc1 = {};
                #pragma unroll
                for (int k = 0; k < 8; ++k) {
                    const int kb = k * 32;
                    // B fragment: W rows are K-contiguous -> one 32B span
                    const v16bf bm = *(const v16bf*)(W + (n0 + laneM) * KDIM
                                                       + kb + 16 * laneH);
                    // A fragments (contiguous K, matching B): two 16B LDS loads each
                    const unsigned short* r0 = xh + (0  + laneM) * XHP + kb;
                    const unsigned short* r1 = xh + (16 + laneM) * XHP + kb;
                    const v16bf a0 = load_frag16(r0 + 16 * laneH, r0 + 16 * laneH + 8);
                    const v16bf a1 = load_frag16(r1 + 16 * laneH, r1 + 16 * laneH + 8);
                    acc0 = __builtin_amdgcn_wmma_f32_16x16x32_bf16(
                               false, a0, false, bm, (short)0, acc0, false, false);
                    acc1 = __builtin_amdgcn_wmma_f32_16x16x32_bf16(
                               false, a1, false, bm, (short)0, acc1, false, false);
                }
                #pragma unroll
                for (int v = 0; v < 8; ++v) {
                    int M = 8 * laneH + v;
                    gates[(M)      * GP + n0 + laneM] = acc0[v];
                    gates[(16 + M) * GP + n0 + laneM] = acc1[v];
                }
            }
            __syncthreads();

            // --- elementwise LSTM cell ---
            float* cs = layer ? c1 : c0;
            for (int e = tid; e < BT * HID; e += 256) {
                int b = e >> 7, j = e & 127;
                float i_ = gates[b * GP + j];
                float f_ = gates[b * GP + HID + j];
                float g_ = gates[b * GP + 2 * HID + j];
                float o_ = gates[b * GP + 3 * HID + j];
                float c  = sigf(f_) * cs[e] + sigf(i_) * tanhf(g_);
                float h  = sigf(o_) * tanhf(c);
                cs[e] = c;
                unsigned short hb = f2bf(h);
                if (layer == 0) {
                    xh0[b * XHP + HID + j] = hb;  // recurrent h0 (next step)
                    xh1[b * XHP + j]       = hb;  // feeds layer1 this step
                } else {
                    xh1[b * XHP + HID + j] = hb;  // recurrent h1 (next step)
                }
            }
            __syncthreads();
        }

        // =============== logit head: TANH_C * tanh((h1 . soft_w)/TEMP) ===============
        const float* SW = P.soft[fi];
        for (int p = tid; p < BT * t; p += 256) {
            int b = p / t, col = p % t;
            float acc = 0.0f;
            #pragma unroll 4
            for (int j = 0; j < HID; ++j)
                acc += bf2f(xh1[b * XHP + HID + j]) * SW[col * HID + j];
            logits[b * 32 + col] = 2.5f * tanhf(acc * (1.0f / 5.0f));
        }
        __syncthreads();

        // =============== decision, loss, next x ===============
        const float* EW = P.emb[fi];
        if (!isAdj) {
            // ---- label field: NLL of log_softmax over 5, x = emb row gather ----
            if (tid < BT) {
                int b = tid;
                int dec = nl[(bg0 + b) * 33 + node];
                decl[b] = dec;
                float m = -1e30f;
                for (int k = 0; k < 5; ++k) m = fmaxf(m, logits[b * 32 + k]);
                float s = 0.0f;
                for (int k = 0; k < 5; ++k) s += __expf(logits[b * 32 + k] - m);
                float lse = m + __logf(s);
                rowAcc[b] += -(logits[b * 32 + dec] - lse);
            }
            __syncthreads();
            for (int e = tid; e < BT * HID; e += 256) {
                int b = e >> 7, j = e & 127;
                xh0[b * XHP + j] = f2bf(EW[decl[b] * HID + j]);
            }
        } else {
            // ---- adjacency field ----
            for (int p = tid; p < BT * t; p += 256) {
                int b = p / t, k = p % t;
                decf[b * 32 + k] = adj[(bg0 + b) * 1089 + node * 33 + k];
            }
            __syncthreads();
            if (tid < BT) {
                int b = tid;
                float m = -1e30f;
                for (int k = 0; k < t; ++k)
                    m = fmaxf(m, sigf(logits[b * 32 + k]));
                float s = 0.0f;
                for (int k = 0; k < t; ++k)
                    s += __expf(sigf(logits[b * 32 + k]) - m);
                float lse = m + __logf(s);
                float lp = 0.0f, den = 1.0f;
                for (int k = 0; k < t; ++k) {
                    float d = decf[b * 32 + k];
                    lp  += d * (sigf(logits[b * 32 + k]) - lse);
                    den += d * (float)k;
                }
                rowAcc[b] += -lp;
                dnm[b] = den;
            }
            __syncthreads();
            for (int e = tid; e < BT * HID; e += 256) {
                int b = e >> 7, j = e & 127;
                float s = 0.0f;
                for (int k = 0; k < t; ++k)
                    s += decf[b * 32 + k] * EW[k * HID + j];
                xh0[b * XHP + j] = f2bf(s / dnm[b]);
            }
        }
        __syncthreads();
    }

    // ---- final scalar reduction ----
    if (tid == 0) {
        float s = 0.0f;
        for (int b = 0; b < BT; ++b) s += rowAcc[b];
        atomicAdd(out, s * (1.0f / (4096.0f * 64.0f)));
    }
}

// ---------------------------------------------------------------------------
extern "C" void kernel_launch(void* const* d_in, const int* in_sizes, int n_in,
                              void* d_out, int out_size, void* d_ws, size_t ws_size,
                              hipStream_t stream) {
    const float* adjp  = (const float*)d_in[0];
    const int*   nlp   = (const int*)  d_in[1];
    const float* gembp = (const float*)d_in[2];
    const float* wih0  = (const float*)d_in[3];
    const float* wih1  = (const float*)d_in[4];
    const float* whh0  = (const float*)d_in[5];
    const float* whh1  = (const float*)d_in[6];

    Ptrs P;
    for (int j = 0; j < NFIELDS; ++j) {
        P.soft[j] = (const float*)d_in[7 + j];
        P.emb[j]  = (const float*)d_in[7 + NFIELDS + j];
    }

    unsigned short* Wcat = (unsigned short*)d_ws;   // 512 KB bf16 weights
    float* outp = (float*)d_out;

    // 2*512*256 = 262144 elements -> 1024 blocks of 256
    lstm_prep<<<1024, 256, 0, stream>>>(wih0, wih1, whh0, whh1, Wcat, outp);
    // 4096 / 32 = 128 persistent blocks
    lstm_main<<<128, 256, 0, stream>>>(adjp, nlp, gembp, Wcat, P, outp);
}
